// ArgumentGCN_3_9680856285899
// MI455X (gfx1250) — matrix operand order repaired and checked by reading, not compiled, not run on hardware
//
#include <hip/hip_runtime.h>
#include <stdint.h>

#define B_  64
#define N_  256
#define D_  512
#define E_  10
#define BN_ (B_ * N_)

typedef __bf16 bf16x16 __attribute__((ext_vector_type(16)));
typedef float  f32x8   __attribute__((ext_vector_type(8)));

union FragB { bf16x16 v; uint32_t u[8]; uint4 q[2]; };

static __device__ inline uint16_t f2bf(float f) {
    union { float f; uint32_t u; } c; c.f = f;
    uint32_t u = c.u;
    uint32_t r = u + 0x7FFFu + ((u >> 16) & 1u);   // round-to-nearest-even
    return (uint16_t)(r >> 16);
}

static __device__ inline f32x8 zero8() {
    f32x8 z;
#pragma unroll
    for (int i = 0; i < 8; ++i) z[i] = 0.0f;
    return z;
}

static __device__ inline f32x8 wmma_bf16(const FragB& a, const FragB& b, f32x8 c) {
    return __builtin_amdgcn_wmma_f32_16x16x32_bf16(false, a.v, false, b.v,
                                                   (short)0, c, false, false);
}

static __device__ inline void loadFrag(FragB& f, const uint32_t* p) {
    f.q[0] = *(const uint4*)p;
    f.q[1] = *(const uint4*)(p + 4);
}

// ---------------------------------------------------------------------------
// k_swzW: swizzle weights [nmat][D][D] (row d, col k, f32) into WMMA B-fragment
// tiles: frag(m,nt,kb) = 32 lanes x 8 u32 (1KB), lane-contiguous.
// lane, vgpr v holds bf16 pair W[d = nt*16 + lane&15][k = kb*32 + (v<4 ? 2v : 16+2(v-4)) + (lane>>4)*8 .. +1]
// ---------------------------------------------------------------------------
__global__ void k_swzW(const float* __restrict__ src, uint32_t* __restrict__ dst,
                       int nmat) {
    int idx = blockIdx.x * blockDim.x + threadIdx.x;   // ((m*32+nt)*16+kb)*32+lane
    if (idx >= (nmat << 14)) return;
    int lane = idx & 31;
    int kb   = (idx >> 5) & 15;
    int nt   = (idx >> 9) & 31;
    int m    = idx >> 14;
    int half = lane >> 4, ln = lane & 15;
    int d = nt * 16 + ln;
    const float* S = src + (size_t)m * D_ * D_ + (size_t)d * D_;
    uint32_t* O = dst + (size_t)idx * 8;
#pragma unroll
    for (int v = 0; v < 8; ++v) {
        int k = kb * 32 + (v < 4 ? 2 * v : 16 + 2 * (v - 4)) + half * 8;
        uint32_t lo = f2bf(S[k]);
        uint32_t hi = f2bf(S[k + 1]);
        O[v] = lo | (hi << 16);
    }
}

// ---------------------------------------------------------------------------
// k_packnw: pack nw (bf16 [BN][D]) into B-fragment tiles frag(b,nt,mb).
// ---------------------------------------------------------------------------
__global__ void k_packnw(const uint16_t* __restrict__ nwb, uint32_t* __restrict__ dst) {
    int idx = blockIdx.x * blockDim.x + threadIdx.x;   // ((b*32+nt)*8+mb)*32+lane
    if (idx >= (B_ << 13)) return;
    int lane = idx & 31;
    int mb   = (idx >> 5) & 7;
    int nt   = (idx >> 8) & 31;
    int b    = idx >> 13;
    int half = lane >> 4, ln = lane & 15;
    int d = nt * 16 + ln;
    const uint16_t* S = nwb + (size_t)b * N_ * D_;
    uint32_t* O = dst + (size_t)idx * 8;
#pragma unroll
    for (int v = 0; v < 8; ++v) {
        int m = mb * 32 + (v < 4 ? 2 * v : 16 + 2 * (v - 4)) + half * 8;
        uint32_t lo = S[(size_t)m * D_ + d];
        uint32_t hi = S[(size_t)(m + 1) * D_ + d];
        O[v] = lo | (hi << 16);
    }
}

// ---------------------------------------------------------------------------
// k_bits: pack masked adjacency into bitmask words + neighbor counts
// ---------------------------------------------------------------------------
__global__ void k_bits(const int* __restrict__ adj, const int* __restrict__ mask,
                       uint32_t* __restrict__ bits, int* __restrict__ ncnt) {
    int idx = blockIdx.x * blockDim.x + threadIdx.x;
    if (idx >= E_ * B_ * N_ * 8) return;
    int wb = idx & 7;
    int n  = (idx >> 3) & 255;
    int b  = (idx >> 11) & 63;
    int e  = idx >> 17;
    int mrow = mask[b * N_ + n];
    const int* arow = adj + ((((size_t)e * B_ + b) * N_ + n) * N_) + wb * 32;
    const int* mcol = mask + b * N_;
    uint32_t w = 0; int cnt = 0;
    if (mrow) {
#pragma unroll
        for (int j = 0; j < 32; ++j) {
            int m = wb * 32 + j;
            int bit = (arow[j] != 0) && (mcol[m] != 0) && (m != n);
            w |= ((uint32_t)bit) << j;
            cnt += bit;
        }
    }
    bits[idx] = w;
    if (cnt) atomicAdd(&ncnt[b * N_ + n], cnt);
}

__global__ void k_inv(const int* __restrict__ ncnt, float* __restrict__ inv) {
    int i = blockIdx.x * blockDim.x + threadIdx.x;
    if (i < BN_) {
        int c = ncnt[i];
        inv[i] = 1.0f / (float)(c < 1 ? 1 : c);
    }
}

// ---------------------------------------------------------------------------
// k_gate: w = sigmoid(node . W_nw + b_nw); emit w, node_bf16, (w*node)_bf16
// ---------------------------------------------------------------------------
__global__ void __launch_bounds__(256)
k_gate(const float* __restrict__ node, const float* __restrict__ Wnw,
       const float* __restrict__ bnw, float* __restrict__ wout, int t,
       uint16_t* __restrict__ nodeb, uint16_t* __restrict__ nwb) {
    int gwave = (blockIdx.x * blockDim.x + threadIdx.x) >> 5;
    int lane  = threadIdx.x & 31;
    if (gwave >= BN_) return;
    const float* row = node + (size_t)gwave * D_;
    float s = 0.0f;
#pragma unroll
    for (int i = 0; i < 16; ++i) s += row[lane + 32 * i] * Wnw[lane + 32 * i];
    for (int off = 16; off > 0; off >>= 1) s += __shfl_down(s, off, 32);
    s = __shfl(s, 0, 32);
    float w = 1.0f / (1.0f + __expf(-(s + bnw[0])));
    if (lane == 0) {
        int b = gwave >> 8, n = gwave & 255;
        wout[(b * 2 + t) * 256 + n] = w;      // all_weight[b][t][n]
    }
    uint16_t* nb = nodeb + (size_t)gwave * D_;
    uint16_t* nw = nwb  + (size_t)gwave * D_;
#pragma unroll
    for (int i = 0; i < 16; ++i) {
        float v = row[lane + 32 * i];
        nb[lane + 32 * i] = f2bf(v);
        nw[lane + 32 * i] = f2bf(v * w);
    }
}

// ---------------------------------------------------------------------------
// k_self: self_info = node @ W_self^T + b_self.
// M-tile 64, 16 waves x 32-wide d-slice, dynamic LDS (64KB A-tile).
// ---------------------------------------------------------------------------
__global__ void __launch_bounds__(512)
k_self(const uint16_t* __restrict__ nodeb, const uint32_t* __restrict__ wsswz,
       const float* __restrict__ bself, float* __restrict__ selfinfo) {
    extern __shared__ char smem[];
    uint16_t* sA = (uint16_t*)smem;            // 64 x 512 bf16 = 64KB
    int row0 = blockIdx.x * 64;
    int tid  = threadIdx.x;
    {
        const uint4* src4 = (const uint4*)(nodeb + (size_t)row0 * D_);
        uint4* dA4 = (uint4*)sA;
        for (int i = tid; i < 64 * D_ / 8; i += 512) dA4[i] = src4[i];
    }
    __syncthreads();
    int wave = tid >> 5, lane = tid & 31;
    int half = lane >> 4, ln = lane & 15;
    int dbase = wave * 32;
    int nt0   = wave * 2;
    f32x8 acc[4][2];
#pragma unroll
    for (int g = 0; g < 4; ++g)
#pragma unroll
        for (int f = 0; f < 2; ++f) acc[g][f] = zero8();

    for (int kb = 0; kb < 16; ++kb) {
        FragB a[4];
#pragma unroll
        for (int g = 0; g < 4; ++g)
#pragma unroll
            for (int j = 0; j < 8; ++j) {
                int klo = (j < 4 ? 2 * j : 16 + 2 * (j - 4)) + half * 8 + kb * 32;
                a[g].u[j] = *(const uint32_t*)&sA[(g * 16 + ln) * D_ + klo];
            }
#pragma unroll
        for (int f = 0; f < 2; ++f) {
            const uint32_t* p = wsswz + (((size_t)(nt0 + f) * 16 + kb) << 8) + lane * 8;
            FragB bf; loadFrag(bf, p);
#pragma unroll
            for (int g = 0; g < 4; ++g) acc[g][f] = wmma_bf16(a[g], bf, acc[g][f]);
        }
    }
#pragma unroll
    for (int g = 0; g < 4; ++g)
#pragma unroll
        for (int f = 0; f < 2; ++f) {
            int dcol = dbase + f * 16 + ln;
            float bs = bself[dcol];
#pragma unroll
            for (int r = 0; r < 8; ++r) {
                int row = row0 + g * 16 + r + half * 8;
                selfinfo[(size_t)row * D_ + dcol] = acc[g][f][r] + bs;
            }
        }
}

// ---------------------------------------------------------------------------
// k_agg: per (b, 64-row n-tile):
//   acc2 = Σ_e ( A_e(bits) @ nw[b] ) @ W_edge[e]^T   (bf16 WMMA, swizzled B)
//   out  = relu(self_info + acc2 * inv_neigh)
// 16 waves x 32-wide d-slice; dynamic LDS: 64KB S-tile + 20KB bits.
// ---------------------------------------------------------------------------
__global__ void __launch_bounds__(512)
k_agg(const uint32_t* __restrict__ bits, const uint32_t* __restrict__ nwswz,
      const uint32_t* __restrict__ weswz, const float* __restrict__ selfinfo,
      const float* __restrict__ inv, float* __restrict__ nodeOut) {
    extern __shared__ char smem[];
    uint16_t* sS    = (uint16_t*)smem;               // 64 x 512 bf16 = 64KB
    uint32_t* sbits = (uint32_t*)(smem + 64 * 1024); // E x 64 x 8 u32 = 20KB
    int b  = blockIdx.x >> 2;
    int n0 = (blockIdx.x & 3) * 64;
    int tid = threadIdx.x;
    for (int i = tid; i < E_ * 64 * 8; i += 512) {
        int wb = i & 7, r = (i >> 3) & 63, e = i >> 9;
        sbits[i] = bits[((((size_t)e * B_ + b) * N_) + n0 + r) * 8 + wb];
    }
    __syncthreads();
    int wave = tid >> 5, lane = tid & 31;
    int half = lane >> 4, ln = lane & 15;
    int dbase = wave * 32;
    int nt0   = wave * 2;
    const uint32_t* nw_b = nwswz + ((size_t)b << 16);         // b * 32*8*256
    f32x8 acc2[4][2];
#pragma unroll
    for (int g = 0; g < 4; ++g)
#pragma unroll
        for (int f = 0; f < 2; ++f) acc2[g][f] = zero8();

    for (int e = 0; e < E_; ++e) {
        const uint32_t* we_e = weswz + ((size_t)e << 17);     // e * 32*16*256
#pragma unroll
        for (int f = 0; f < 2; ++f)   // warm W_edge stream for this e (global_prefetch)
            __builtin_prefetch((const void*)(we_e + (((size_t)(nt0 + f) * 16) << 8) + lane * 8), 0, 0);

        // -------- GEMM1: S[:, my 32 cols] = A_e @ nw[b] --------
        f32x8 acc1[4][2];
#pragma unroll
        for (int g = 0; g < 4; ++g)
#pragma unroll
            for (int f = 0; f < 2; ++f) acc1[g][f] = zero8();
        for (int mb = 0; mb < 8; ++mb) {
            FragB a[4];
#pragma unroll
            for (int g = 0; g < 4; ++g) {
                uint32_t w = sbits[(e * 64 + g * 16 + ln) * 8 + mb];
#pragma unroll
                for (int j = 0; j < 8; ++j) {
                    int klo = (j < 4 ? 2 * j : 16 + 2 * (j - 4)) + half * 8;
                    uint32_t fld = (w >> klo) & 3u;
                    a[g].u[j] = (fld & 1u) * 0x3F80u + (fld >> 1) * 0x3F800000u;
                }
            }
#pragma unroll
            for (int f = 0; f < 2; ++f) {
                const uint32_t* p = nw_b + (((size_t)(nt0 + f) * 8 + mb) << 8) + lane * 8;
                FragB bf; loadFrag(bf, p);
#pragma unroll
                for (int g = 0; g < 4; ++g) acc1[g][f] = wmma_bf16(a[g], bf, acc1[g][f]);
            }
        }
        __syncthreads();   // previous GEMM2 consumers of sS are done
#pragma unroll
        for (int g = 0; g < 4; ++g)
#pragma unroll
            for (int f = 0; f < 2; ++f) {
                int dcol = dbase + f * 16 + ln;
#pragma unroll
                for (int r = 0; r < 8; ++r)
                    sS[(g * 16 + r + half * 8) * D_ + dcol] = f2bf(acc1[g][f][r]);
            }
        __syncthreads();   // sS ready
        // -------- GEMM2: acc2 += S @ W_edge[e]^T --------
        for (int kb = 0; kb < 16; ++kb) {
            FragB a[4];
#pragma unroll
            for (int g = 0; g < 4; ++g)
#pragma unroll
                for (int j = 0; j < 8; ++j) {
                    int klo = (j < 4 ? 2 * j : 16 + 2 * (j - 4)) + half * 8 + kb * 32;
                    a[g].u[j] = *(const uint32_t*)&sS[(g * 16 + ln) * D_ + klo];
                }
#pragma unroll
            for (int f = 0; f < 2; ++f) {
                const uint32_t* p = we_e + (((size_t)(nt0 + f) * 16 + kb) << 8) + lane * 8;
                FragB bf; loadFrag(bf, p);
#pragma unroll
                for (int g = 0; g < 4; ++g) acc2[g][f] = wmma_bf16(a[g], bf, acc2[g][f]);
            }
        }
    }
    // -------- epilogue: relu(self + agg * inv_neigh) --------
#pragma unroll
    for (int g = 0; g < 4; ++g)
#pragma unroll
        for (int f = 0; f < 2; ++f) {
            int dcol = dbase + f * 16 + ln;
#pragma unroll
            for (int r = 0; r < 8; ++r) {
                int n = n0 + g * 16 + r + half * 8;
                size_t gi = ((size_t)b * N_ + n) * D_ + dcol;
                float v = selfinfo[gi] + acc2[g][f][r] * inv[b * N_ + n];
                nodeOut[gi] = v > 0.0f ? v : 0.0f;
            }
        }
}

// ---------------------------------------------------------------------------
extern "C" void kernel_launch(void* const* d_in, const int* in_sizes, int n_in,
                              void* d_out, int out_size, void* d_ws, size_t ws_size,
                              hipStream_t stream) {
    const float* node0 = (const float*)d_in[0];
    const float* Wnw   = (const float*)d_in[1];
    const float* bnw   = (const float*)d_in[2];
    const float* Wself = (const float*)d_in[3];
    const float* bself = (const float*)d_in[4];
    const float* Wedge = (const float*)d_in[5];
    const int*   mask  = (const int*)d_in[6];
    const int*   adj   = (const int*)d_in[7];
    float* outNode = (float*)d_out;
    float* outW    = outNode + (size_t)BN_ * D_;

    char* ws = (char*)d_ws;
    size_t off = 0;
    auto alloc = [&](size_t bytes) -> void* {
        void* p = ws + off; off += (bytes + 255) & ~(size_t)255; return p;
    };
    uint16_t* nodeb    = (uint16_t*)alloc((size_t)BN_ * D_ * 2);
    uint16_t* nwb      = (uint16_t*)alloc((size_t)BN_ * D_ * 2);
    float*    selfinfo = (float*)   alloc((size_t)BN_ * D_ * 4);
    float*    nodecur  = (float*)   alloc((size_t)BN_ * D_ * 4);
    uint32_t* wsswz    = (uint32_t*)alloc((size_t)(1 << 14) * 32);        // 0.5 MB
    uint32_t* weswz    = (uint32_t*)alloc((size_t)(E_ << 14) * 32);       // 5.2 MB
    uint32_t* nwswz    = (uint32_t*)alloc((size_t)(B_ << 13) * 32);       // 16.8 MB
    uint32_t* bitsbuf  = (uint32_t*)alloc((size_t)E_ * B_ * N_ * 8 * 4);  // 5.2 MB
    int*      ncnt     = (int*)     alloc((size_t)BN_ * 4);
    float*    invbuf   = (float*)   alloc((size_t)BN_ * 4);

    hipMemsetAsync(ncnt, 0, (size_t)BN_ * 4, stream);
    k_swzW<<<((1 << 14) + 255) / 256, 256, 0, stream>>>(Wself, wsswz, 1);
    k_swzW<<<((E_ << 14) + 255) / 256, 256, 0, stream>>>(Wedge, weswz, E_);
    k_bits<<<(E_ * B_ * N_ * 8 + 255) / 256, 256, 0, stream>>>(adj, mask, bitsbuf, ncnt);
    k_inv <<<(BN_ + 255) / 256, 256, 0, stream>>>(ncnt, invbuf);

    const size_t lds_self = 64 * 1024;
    const size_t lds_agg  = 64 * 1024 + E_ * 64 * 8 * 4;   // 84 KB

    const float* cur = node0;
    for (int t = 0; t < 2; ++t) {
        k_gate  <<<BN_ / 8, 256, 0, stream>>>(cur, Wnw, bnw, outW, t, nodeb, nwb);
        k_packnw<<<(B_ << 13) / 256, 256, 0, stream>>>(nwb, nwswz);
        k_self  <<<BN_ / 64, 512, lds_self, stream>>>(nodeb, wsswz, bself, selfinfo);
        float* nxt = (t == 1) ? outNode : nodecur;
        k_agg   <<<B_ * 4, 512, lds_agg, stream>>>(bitsbuf, nwswz, weswz, selfinfo, invbuf, nxt);
        cur = nxt;
    }
}
